// transPPI_share_43714177139194
// MI455X (gfx1250) — matrix operand order repaired and checked by reading, not compiled, not run on hardware
//
#include <hip/hip_runtime.h>

// ---------------------------------------------------------------------------
// Types for CDNA5 WMMA (wave32, 16x16x32 bf16 -> f32)
// ---------------------------------------------------------------------------
typedef __attribute__((ext_vector_type(16))) __bf16 bf16x16;
typedef __attribute__((ext_vector_type(8)))  __bf16 bf16x8;
typedef __attribute__((ext_vector_type(8)))  float  f32x8;
typedef __attribute__((ext_vector_type(8)))  unsigned short u16x8;

__device__ __forceinline__ unsigned short f32_to_bf16_rne(float f) {
  unsigned u = __float_as_uint(f);
  unsigned r = u + 0x7FFFu + ((u >> 16) & 1u);   // round-to-nearest-even
  return (unsigned short)(r >> 16);
}

// ---------------------------------------------------------------------------
// Generic GEMM:  C[M,Nc] = act( A[M,K] @ W[K,Nc] + bias )
//   A is f32 (optionally a K-split concat of A0[:,0:ksplit] and A1),
//   W is pre-converted bf16, TRANSPOSED layout Wt[n*K + k].
// Block = 256 threads (8 waves). Block computes 32 rows x Nc cols:
//   2 row-tiles shared by all waves, NT col-tiles per wave (NT = Nc/128,
//   templated so the WMMA loop is guard-free straight-line code).
// Stage = 32 rows x 64 K of bf16 A in LDS per barrier pair
//   -> 8 WMMAs per wave per stage (NT=2), 4 independent acc chains.
// ---------------------------------------------------------------------------
template <int NT>
__global__ __launch_bounds__(256) void gemm_bf16_wmma(
    const float* __restrict__ A0, const float* __restrict__ A1, int ksplit,
    const unsigned short* __restrict__ Wt_u16, const float* __restrict__ bias,
    float* __restrict__ C, int M, int Nc, int K, int act /*0:none 1:relu*/)
{
  // 32 rows x 64 k, pitch 72 elems (144 B): 16B-aligned rows, and the
  // b128 fragment reads hit distinct banks (36*l mod 64 distinct, l=0..15).
  __shared__ __align__(16) unsigned short As[32 * 72];
  const int tid    = threadIdx.x;
  const int wave   = tid >> 5;
  const int lane   = tid & 31;
  const int laneHi = lane >> 4;      // 0|1
  const int laneLo = lane & 15;
  const int rowBase = blockIdx.x * 32;
  const __bf16* Wt = reinterpret_cast<const __bf16*>(Wt_u16);

  f32x8 acc[2][NT];
#pragma unroll
  for (int rt = 0; rt < 2; ++rt)
#pragma unroll
    for (int nt = 0; nt < NT; ++nt)
      acc[rt][nt] = (f32x8){0.f,0.f,0.f,0.f,0.f,0.f,0.f,0.f};

  // staging map: thread -> (row, 8 consecutive k)
  const int sr = tid >> 3;          // 0..31
  const int sk = (tid & 7) * 8;     // 0..56

  for (int kb = 0; kb < K; kb += 64) {
    // Select concat side (ksplit is a multiple of 64 -> no straddling)
    const float* Ap; int lda, kloc;
    if (kb < ksplit) { Ap = A0; lda = ksplit;     kloc = kb; }
    else             { Ap = A1; lda = K - ksplit; kloc = kb - ksplit; }

    // Stage A: two float4 loads -> 8x bf16 -> one 16B LDS store
    {
      int grow = rowBase + sr;
      float4 v0 = {0.f,0.f,0.f,0.f}, v1 = v0;
      if (grow < M) {
        const float4* p =
            reinterpret_cast<const float4*>(Ap + (size_t)grow * lda + kloc + sk);
        v0 = p[0]; v1 = p[1];
      }
      u16x8 pk;
      pk[0] = f32_to_bf16_rne(v0.x); pk[1] = f32_to_bf16_rne(v0.y);
      pk[2] = f32_to_bf16_rne(v0.z); pk[3] = f32_to_bf16_rne(v0.w);
      pk[4] = f32_to_bf16_rne(v1.x); pk[5] = f32_to_bf16_rne(v1.y);
      pk[6] = f32_to_bf16_rne(v1.z); pk[7] = f32_to_bf16_rne(v1.w);
      *reinterpret_cast<u16x8*>(&As[sr * 72 + sk]) = pk;
    }
    __syncthreads();

#pragma unroll
    for (int kk = 0; kk < 64; kk += 32) {
      // A fragments (16-bit A 16x32 layout):
      //  lanes 0-15 : row=laneLo,  K {0..7} in h[0],  K {16..23} in h[1]
      //  lanes16-31 : row=laneLo,  K {8..15} in h[0], K {24..31} in h[1]
      union { bf16x16 v; bf16x8 h[2]; } af[2];
#pragma unroll
      for (int rt = 0; rt < 2; ++rt) {
        const unsigned short* base =
            &As[(rt * 16 + laneLo) * 72 + kk + laneHi * 8];
        af[rt].h[0] = *reinterpret_cast<const bf16x8*>(base);
        af[rt].h[1] = *reinterpret_cast<const bf16x8*>(base + 16);
      }
      // B fragments from transposed bf16 weights (col-contiguous K),
      // reused across both row tiles.
#pragma unroll
      for (int nt = 0; nt < NT; ++nt) {
        int col = (wave + nt * 8) * 16 + laneLo;
        const __bf16* bp = Wt + (size_t)col * K + kb + kk + laneHi * 8;
        union { bf16x16 v; bf16x8 h[2]; } bfr;
        bfr.h[0] = *reinterpret_cast<const bf16x8*>(bp);
        bfr.h[1] = *reinterpret_cast<const bf16x8*>(bp + 16);
#pragma unroll
        for (int rt = 0; rt < 2; ++rt)
          acc[rt][nt] = __builtin_amdgcn_wmma_f32_16x16x32_bf16(
              false, af[rt].v, false, bfr.v, (short)0, acc[rt][nt], false, false);
      }
    }
    __syncthreads();
  }

  // C layout: lane owns col (laneLo), rows r + 8*laneHi in VGPRs 0..7
#pragma unroll
  for (int nt = 0; nt < NT; ++nt) {
    int col = (wave + nt * 8) * 16 + laneLo;
    float bv = bias ? bias[col] : 0.f;
#pragma unroll
    for (int rt = 0; rt < 2; ++rt) {
#pragma unroll
      for (int r = 0; r < 8; ++r) {
        int row = rowBase + rt * 16 + r + laneHi * 8;
        if (row < M) {
          float v = acc[rt][nt][r] + bv;
          if (act == 1) v = fmaxf(v, 0.f);
          C[(size_t)row * Nc + col] = v;
        }
      }
    }
  }
}

// ---------------------------------------------------------------------------
// COO SpMM scatter: Y[dst] += val * X[src], F=256.  One wave per edge.
// X and Y (51 MB each) are L2-resident on MI455X (192 MB L2).
// ---------------------------------------------------------------------------
__global__ __launch_bounds__(256) void spmm_scatter256(
    const int* __restrict__ src, const int* __restrict__ dst,
    const float* __restrict__ val,
    const float* __restrict__ X, float* __restrict__ Y, int E)
{
  int e = blockIdx.x * 8 + (threadIdx.x >> 5);
  if (e >= E) return;
  int lane = threadIdx.x & 31;
  int s = src[e], d = dst[e];
  float v = val[e];
  const float4* xs = reinterpret_cast<const float4*>(X + (size_t)s * 256);
  float* yd = Y + (size_t)d * 256;
#pragma unroll
  for (int it = 0; it < 2; ++it) {
    int c4 = lane + it * 32;
    float4 x = xs[c4];
    atomicAdd(yd + c4 * 4 + 0, v * x.x);
    atomicAdd(yd + c4 * 4 + 1, v * x.y);
    atomicAdd(yd + c4 * 4 + 2, v * x.z);
    atomicAdd(yd + c4 * 4 + 3, v * x.w);
  }
}

// ---------------------------------------------------------------------------
// Elementwise helpers
// ---------------------------------------------------------------------------
__global__ void fill_zero_f4(float4* __restrict__ p, long n4) {
  long i = (long)blockIdx.x * blockDim.x + threadIdx.x;
  if (i < n4) p[i] = make_float4(0.f, 0.f, 0.f, 0.f);
}

__global__ void relu_f4(float4* __restrict__ p, long n4) {
  long i = (long)blockIdx.x * blockDim.x + threadIdx.x;
  if (i < n4) {
    float4 v = p[i];
    v.x = fmaxf(v.x, 0.f); v.y = fmaxf(v.y, 0.f);
    v.z = fmaxf(v.z, 0.f); v.w = fmaxf(v.w, 0.f);
    p[i] = v;
  }
}

// out[i] = dot(X[i,:], w)
__global__ void rowdot(const float* __restrict__ X, const float* __restrict__ w,
                       float* __restrict__ out, int M, int Fdim)
{
  int i = blockIdx.x * blockDim.x + threadIdx.x;
  if (i >= M) return;
  const float4* xr = reinterpret_cast<const float4*>(X + (size_t)i * Fdim);
  const float4* wv = reinterpret_cast<const float4*>(w);
  float s = 0.f;
  for (int c = 0; c < (Fdim >> 2); ++c) {
    float4 a = xr[c], b = wv[c];
    s += a.x * b.x + a.y * b.y + a.z * b.z + a.w * b.w;
  }
  out[i] = s;
}

// score[i] = sigmoid(dot(X[i,:], w) + bias[0])
__global__ void rowdot_sigmoid(const float* __restrict__ X, const float* __restrict__ w,
                               const float* __restrict__ bias,
                               float* __restrict__ out, int M, int Fdim)
{
  int i = blockIdx.x * blockDim.x + threadIdx.x;
  if (i >= M) return;
  const float4* xr = reinterpret_cast<const float4*>(X + (size_t)i * Fdim);
  const float4* wv = reinterpret_cast<const float4*>(w);
  float s = 0.f;
  for (int c = 0; c < (Fdim >> 2); ++c) {
    float4 a = xr[c], b = wv[c];
    s += a.x * b.x + a.y * b.y + a.z * b.z + a.w * b.w;
  }
  s += bias[0];
  out[i] = 1.f / (1.f + expf(-s));
}

// Single-block softmax over n elements (n ~ 50k: 3 grid-stride passes)
__global__ __launch_bounds__(1024) void softmax_vec(
    const float* __restrict__ x, float* __restrict__ y, int n)
{
  __shared__ float red[1024];
  int tid = threadIdx.x;
  float m = -3.4e38f;
  for (int i = tid; i < n; i += 1024) m = fmaxf(m, x[i]);
  red[tid] = m; __syncthreads();
  for (int s = 512; s > 0; s >>= 1) {
    if (tid < s) red[tid] = fmaxf(red[tid], red[tid + s]);
    __syncthreads();
  }
  float mx = red[0];
  __syncthreads();
  float acc = 0.f;
  for (int i = tid; i < n; i += 1024) acc += expf(x[i] - mx);
  red[tid] = acc; __syncthreads();
  for (int s = 512; s > 0; s >>= 1) {
    if (tid < s) red[tid] += red[tid + s];
    __syncthreads();
  }
  float inv = 1.f / red[0];
  for (int i = tid; i < n; i += 1024) y[i] = expf(x[i] - mx) * inv;
}

// residue i <- mean of kmer rows [clip(i-2,0,nk), clip(i+1,0,nk)), F=256
__global__ void kmer_recover(const float* __restrict__ qx, float* __restrict__ out,
                             int n_res, int n_kmer)
{
  long idx = (long)blockIdx.x * blockDim.x + threadIdx.x;
  long total = (long)n_res * 256;
  if (idx >= total) return;
  int i = (int)(idx >> 8);
  int c = (int)(idx & 255);
  int lo = i - 2; if (lo < 0) lo = 0; if (lo > n_kmer) lo = n_kmer;
  int hi = i + 1; if (hi > n_kmer) hi = n_kmer;
  int cnt = hi - lo; if (cnt < 1) cnt = 1;
  float s = 0.f;
  for (int j = lo; j < hi; ++j) s += qx[(size_t)j * 256 + c];
  out[(size_t)i * 256 + c] = s / (float)cnt;
}

// W[K,Nc] f32 -> Wt[Nc,K] bf16 (transposed so GEMM B-fragments are contiguous)
__global__ void conv_w_bf16t(const float* __restrict__ W, unsigned short* __restrict__ Wt,
                             int K, int Nc)
{
  int idx = blockIdx.x * blockDim.x + threadIdx.x;
  if (idx >= K * Nc) return;
  int k = idx / Nc, n = idx - k * Nc;
  Wt[(size_t)n * K + k] = f32_to_bf16_rne(W[(size_t)k * Nc + n]);
}

// ---------------------------------------------------------------------------
extern "C" void kernel_launch(void* const* d_in, const int* in_sizes, int n_in,
                              void* d_out, int out_size, void* d_ws, size_t ws_size,
                              hipStream_t stream)
{
  (void)n_in; (void)out_size; (void)ws_size;
  const float* struct_x = (const float*)d_in[0];
  const float* seq_x    = (const float*)d_in[1];
  const int*   src_s = (const int*)d_in[2];
  const int*   dst_s = (const int*)d_in[3];
  const float* val_s = (const float*)d_in[4];
  const int*   src_q = (const int*)d_in[5];
  const int*   dst_q = (const int*)d_in[6];
  const float* val_q = (const float*)d_in[7];
  const float* W1  = (const float*)d_in[8];
  const float* b1  = (const float*)d_in[9];
  const float* gc1 = (const float*)d_in[10];
  const float* gc2 = (const float*)d_in[11];
  const float* gap = (const float*)d_in[12];
  const float* W2  = (const float*)d_in[13];
  const float* b2  = (const float*)d_in[14];
  const float* f1  = (const float*)d_in[15];
  const float* fb1 = (const float*)d_in[16];
  const float* f2  = (const float*)d_in[17];
  const float* fb2 = (const float*)d_in[18];
  const float* f3  = (const float*)d_in[19];
  const float* fb3 = (const float*)d_in[20];

  const int Din = 1024, H = 256;
  const int N  = in_sizes[0] / Din;   // 50000
  const int Nq = in_sizes[1] / Din;   // 49998
  const int E  = in_sizes[2];         // 800000

  float* out       = (float*)d_out;
  float* out_score = out;                       // [N]
  float* out_satt  = out + N;                   // [N]
  float* out_qatt  = out + 2 * (size_t)N;       // [Nq]
  float* out_x     = out + 2 * (size_t)N + Nq;  // [N,256]

  // ---- workspace carve-up -------------------------------------------------
  char* ws = (char*)d_ws;
  size_t off = 0;
  auto take = [&](size_t bytes) -> void* {
    void* p = ws + off;
    off = (off + bytes + 255) & ~(size_t)255;
    return p;
  };
  unsigned short* W1t  = (unsigned short*)take((size_t)Din * H   * 2);
  unsigned short* gc1t = (unsigned short*)take((size_t)H   * 256 * 2);
  unsigned short* gc2t = (unsigned short*)take((size_t)H   * 256 * 2);
  unsigned short* W2t  = (unsigned short*)take((size_t)512 * 256 * 2);
  unsigned short* f1t  = (unsigned short*)take((size_t)256 * 256 * 2);
  unsigned short* f2t  = (unsigned short*)take((size_t)256 * 128 * 2);
  float* Ls = (float*)take((size_t)N  * 4);
  float* Lq = (float*)take((size_t)Nq * 4);
  float* B0 = (float*)take((size_t)N * 256 * 4);
  float* B1 = (float*)take((size_t)N * 256 * 4);
  float* B2 = (float*)take((size_t)N * 256 * 4);

  auto cdiv = [](long a, long b) { return (int)((a + b - 1) / b); };
  const long NF  = (long)N  * 256;
  const long NqF = (long)Nq * 256;

  auto zero = [&](float* p, long n) {
    fill_zero_f4<<<cdiv(n / 4, 256), 256, 0, stream>>>((float4*)p, n / 4);
  };
  auto relu = [&](float* p, long n) {
    relu_f4<<<cdiv(n / 4, 256), 256, 0, stream>>>((float4*)p, n / 4);
  };
  auto gemm = [&](const float* A0, const float* A1, int ksplit,
                  const unsigned short* Wt, const float* bias,
                  float* C, int M, int Nc, int K, int act) {
    if (Nc == 256)
      gemm_bf16_wmma<2><<<cdiv(M, 32), 256, 0, stream>>>(A0, A1, ksplit, Wt,
                                                         bias, C, M, Nc, K, act);
    else  // Nc == 128
      gemm_bf16_wmma<1><<<cdiv(M, 32), 256, 0, stream>>>(A0, A1, ksplit, Wt,
                                                         bias, C, M, Nc, K, act);
  };

  // ---- weight conversion (f32 -> transposed bf16) -------------------------
  conv_w_bf16t<<<cdiv((long)Din * H, 256),  256, 0, stream>>>(W1,  W1t,  Din, H);
  conv_w_bf16t<<<cdiv((long)H * 256, 256),  256, 0, stream>>>(gc1, gc1t, H,   256);
  conv_w_bf16t<<<cdiv((long)H * 256, 256),  256, 0, stream>>>(gc2, gc2t, H,   256);
  conv_w_bf16t<<<cdiv((long)512 * 256, 256),256, 0, stream>>>(W2,  W2t,  512, 256);
  conv_w_bf16t<<<cdiv((long)256 * 256, 256),256, 0, stream>>>(f1,  f1t,  256, 256);
  conv_w_bf16t<<<cdiv((long)256 * 128, 256),256, 0, stream>>>(f2,  f2t,  256, 128);

  // ---- structure branch ---------------------------------------------------
  gemm(struct_x, nullptr, Din, W1t, b1, B0, N, 256, Din, 1);      // h_s = relu(X@W1+b1)
  gemm(B0, nullptr, 256, gc1t, nullptr, B1, N, 256, 256, 0);      // t = h_s@gc1
  zero(B0, NF);
  spmm_scatter256<<<cdiv(E, 8), 256, 0, stream>>>(src_s, dst_s, val_s, B1, B0, E);
  relu(B0, NF);                                                    // sx2 = B0
  gemm(B0, nullptr, 256, gc2t, nullptr, B1, N, 256, 256, 0);      // t2
  zero(B0, NF);
  spmm_scatter256<<<cdiv(E, 8), 256, 0, stream>>>(src_s, dst_s, val_s, B1, B0, E);
  relu(B0, NF);                                                    // sx3 = B0
  rowdot<<<cdiv(N, 256), 256, 0, stream>>>(B0, gap, Ls, N, 256);
  softmax_vec<<<1, 1024, 0, stream>>>(Ls, out_satt, N);

  // ---- sequence (kmer) branch --------------------------------------------
  gemm(seq_x, nullptr, Din, W1t, b1, B1, Nq, 256, Din, 1);        // h_q = B1
  gemm(B1, nullptr, 256, gc1t, nullptr, B2, Nq, 256, 256, 0);     // t
  zero(B1, NqF);
  spmm_scatter256<<<cdiv(E, 8), 256, 0, stream>>>(src_q, dst_q, val_q, B2, B1, E);
  relu(B1, NqF);                                                   // qx2 = B1
  gemm(B1, nullptr, 256, gc2t, nullptr, B2, Nq, 256, 256, 0);     // t2
  zero(B1, NqF);
  spmm_scatter256<<<cdiv(E, 8), 256, 0, stream>>>(src_q, dst_q, val_q, B2, B1, E);
  relu(B1, NqF);                                                   // qx3 = B1
  rowdot<<<cdiv(Nq, 256), 256, 0, stream>>>(B1, gap, Lq, Nq, 256);
  softmax_vec<<<1, 1024, 0, stream>>>(Lq, out_qatt, Nq);

  // ---- residue recovery + fusion -----------------------------------------
  kmer_recover<<<cdiv(NF, 256), 256, 0, stream>>>(B1, B2, N, Nq); // residu = B2
  // x = [sx3 | residu] @ W2 + b2   (K-split concat GEMM, no concat buffer)
  gemm(B0, B2, 256, W2t, b2, out_x, N, 256, 512, 0);

  // ---- MLP head -----------------------------------------------------------
  gemm(out_x, nullptr, 256, f1t, fb1, B1, N, 256, 256, 1);        // hf1
  gemm(B1,    nullptr, 256, f2t, fb2, B2, N, 128, 256, 1);        // hf2
  rowdot_sigmoid<<<cdiv(N, 256), 256, 0, stream>>>(B2, f3, fb3, out_score, N, 128);
}
